// SinglePrediction_88493506167138
// MI455X (gfx1250) — matrix operand order repaired and checked by reading, not compiled, not run on hardware
//
#include <hip/hip_runtime.h>
#include <hip/hip_bf16.h>
#include <stdint.h>

typedef __attribute__((ext_vector_type(16))) __bf16 v16bf;
typedef __attribute__((ext_vector_type(8)))  float  v8f;

namespace {
constexpr int HID   = 64;
constexpr int GH    = 4 * HID;   // 256 gate columns (i,f,g,o blocks of 64)
constexpr int T_ENC = 20;
constexpr int T_DEC = 30;
constexpr int WS    = 72;        // padded LDS row stride in bf16 elements
constexpr int ROWS_PER_WAVE  = 16;   // one 16-row WMMA A-tile per wave
constexpr int WAVES          = 8;
constexpr int ROWS_PER_BLOCK = ROWS_PER_WAVE * WAVES; // 128
}

// Fast activations: v_exp_f32 + v_rcp_f32 (no IEEE division sequences)
__device__ __forceinline__ float fast_rcp(float x)     { return __builtin_amdgcn_rcpf(x); }
__device__ __forceinline__ float fast_sigmoid(float x) { return fast_rcp(1.0f + __expf(-x)); }
__device__ __forceinline__ float fast_tanh(float x)    { return 1.0f - 2.0f * fast_rcp(1.0f + __expf(2.0f * x)); }

#define WMMA_BF16(A, B, C) \
  __builtin_amdgcn_wmma_f32_16x16x32_bf16(false, (A), false, (B), (short)0, (C), false, false)

union FragU { uint4 q[2]; v16bf v; };

// B fragment (K=32 slab): lane owns gate column `col`; halves 0..7 = K klo+hh*8.., halves 8..15 = K klo+16+hh*8..
__device__ __forceinline__ v16bf ld_bfrag(const __bf16* sW, int col, int klo, int hh) {
  const __bf16* p = sW + col * WS + klo + hh * 8;
  FragU f;
  f.q[0] = *(const uint4*)(p);
  f.q[1] = *(const uint4*)(p + 16);
  return f.v;
}

// A fragment (16x32 bf16): lane row = rl; halves 0..7 = K c0..c0+7, halves 8..15 = K c0+16..
__device__ __forceinline__ v16bf ld_afrag(const __bf16* hrow, int klo, int hh) {
  const __bf16* p = hrow + klo + hh * 8;
  FragU f;
  f.q[0] = *(const uint4*)(p);
  f.q[1] = *(const uint4*)(p + 16);
  return f.v;
}

__global__ __launch_bounds__(256, 4)   // waves-per-eu>=4 -> <=256 arch VGPRs, no MSB toggles
void lstm_seq2seq_wmma_kernel(const float* __restrict__ traj,
                              const float* __restrict__ Wih_enc,
                              const float* __restrict__ Whh_enc,
                              const float* __restrict__ bih_enc,
                              const float* __restrict__ bhh_enc,
                              const float* __restrict__ Wih_dec,
                              const float* __restrict__ Whh_dec,
                              const float* __restrict__ bih_dec,
                              const float* __restrict__ bhh_dec,
                              const float* __restrict__ Wemb,
                              const float* __restrict__ bemb,
                              float* __restrict__ out) {
  __shared__ __attribute__((aligned(16))) __bf16 sWenc[GH * WS];  // Whh_enc, row-major [col][k]
  __shared__ __attribute__((aligned(16))) __bf16 sWdec[GH * WS];  // Wih_dec + Whh_dec combined
  __shared__ __attribute__((aligned(16))) __bf16 sWih [GH * 2];   // encoder input weights (K=2)
  __shared__ float  sBenc[GH];
  __shared__ float  sBdec[GH];
  __shared__ float  sWemb[2 * HID];
  __shared__ float  sBemb[2];
  __shared__ __attribute__((aligned(16))) __bf16 sH[WAVES * ROWS_PER_WAVE * WS];

  const int tid  = threadIdx.x;
  const int lane = tid & 31;
  const int wave = tid >> 5;
  const int rl   = lane & 15;   // row/col index inside a 16-wide tile
  const int hh   = lane >> 4;   // upper half-wave flag

  // ---- stage weights into LDS (bf16), fold decoder x==h and fuse biases ----
  for (int i = tid; i < GH * HID; i += 256) {
    int col = i >> 6, k = i & 63;
    sWenc[col * WS + k] = (__bf16)Whh_enc[i];
    sWdec[col * WS + k] = (__bf16)(Wih_dec[i] + Whh_dec[i]);
  }
  for (int i = tid; i < GH * 2; i += 256) sWih[i] = (__bf16)Wih_enc[i];
  for (int i = tid; i < GH; i += 256) {
    sBenc[i] = bih_enc[i] + bhh_enc[i];
    sBdec[i] = bih_dec[i] + bhh_dec[i];
  }
  for (int i = tid; i < 2 * HID; i += 256) sWemb[i] = Wemb[i];
  if (tid < 2) sBemb[tid] = bemb[tid];
  for (int i = tid; i < WAVES * ROWS_PER_WAVE * WS; i += 256) sH[i] = (__bf16)0.0f;
  __syncthreads();

  const int rowBase = blockIdx.x * ROWS_PER_BLOCK + wave * ROWS_PER_WAVE;
  __bf16* hbuf = &sH[wave * ROWS_PER_WAVE * WS];   // private to this wave: no barriers needed
  const __bf16* harow = &hbuf[rl * WS];            // this lane's A-fragment row

  // constant zero accumulator (SRC2 of the first WMMA in each chain -> inline 0)
  v8f zacc;
  #pragma unroll
  for (int e = 0; e < 8; e++) zacc[e] = 0.0f;

  v8f cst[4];   // cell state, D-layout per hidden-col group
  #pragma unroll
  for (int ng = 0; ng < 4; ng++)
    #pragma unroll
    for (int e = 0; e < 8; e++) cst[ng][e] = 0.0f;

  // x A-fragment: only elements 0,1 ever change (lanes<16); rest stay zero
  v16bf xf;
  #pragma unroll
  for (int i = 0; i < 16; i++) xf[i] = (__bf16)0.0f;

  // Wih B-fragment: only elements 0,1 change per tile (lanes<16); rest stay zero
  v16bf bw;
  #pragma unroll
  for (int i = 0; i < 16; i++) bw[i] = (__bf16)0.0f;

  // ================= encoder: 20 steps =================
  #pragma unroll 1
  for (int t = 0; t < T_ENC; t++) {
    v16bf ha0 = ld_afrag(harow, 0,  hh);
    v16bf ha1 = ld_afrag(harow, 32, hh);

    if (hh == 0) {
      const float* xp = traj + ((size_t)(rowBase + rl) * T_ENC + t) * 2;
      xf[0] = (__bf16)xp[0];
      xf[1] = (__bf16)xp[1];
    }

    // per-gate accumulator chain: D = x*Wih + h*Whh (bias folded into activation)
    auto enc_gate = [&](int g, int ng, v8f& acc) -> float {
      const int col = (g * 4 + ng) * 16 + rl;
      if (hh == 0) { bw[0] = sWih[col * 2]; bw[1] = sWih[col * 2 + 1]; }
      const v16bf b0 = ld_bfrag(sWenc, col, 0,  hh);
      const v16bf b1 = ld_bfrag(sWenc, col, 32, hh);
      v8f a = WMMA_BF16(xf,  bw, zacc);
      a     = WMMA_BF16(ha0, b0, a);
      a     = WMMA_BF16(ha1, b1, a);
      acc = a;
      return sBenc[col];
    };

    #pragma unroll
    for (int ng = 0; ng < 4; ng++) {
      // stage 1: input gate (i) and candidate (g) -> ig
      v8f ai, ag;
      const float bi = enc_gate(0, ng, ai);
      const float bg = enc_gate(2, ng, ag);
      v8f ig;
      #pragma unroll
      for (int e = 0; e < 8; e++)
        ig[e] = fast_sigmoid(ai[e] + bi) * fast_tanh(ag[e] + bg);

      // stage 2: forget gate -> cell update
      v8f af;
      const float bf_ = enc_gate(1, ng, af);
      #pragma unroll
      for (int e = 0; e < 8; e++)
        cst[ng][e] = fast_sigmoid(af[e] + bf_) * cst[ng][e] + ig[e];

      // stage 3: output gate -> h, store to LDS (bf16, A-load layout)
      v8f ao;
      const float bo = enc_gate(3, ng, ao);
      #pragma unroll
      for (int e = 0; e < 8; e++) {
        const float hn = fast_sigmoid(ao[e] + bo) * fast_tanh(cst[ng][e]);
        hbuf[(e + hh * 8) * WS + ng * 16 + rl] = (__bf16)hn;
      }
      // keep scheduler from pipelining across hidden-col groups (bounds VGPR pressure)
      __builtin_amdgcn_sched_barrier(0);
    }
  }

  // ================= decoder: 30 steps =================
  #pragma unroll 1
  for (int t = 0; t < T_DEC; t++) {
    v16bf ha0 = ld_afrag(harow, 0,  hh);
    v16bf ha1 = ld_afrag(harow, 32, hh);

    auto dec_gate = [&](int g, int ng, v8f& acc) -> float {
      const int col = (g * 4 + ng) * 16 + rl;
      const v16bf b0 = ld_bfrag(sWdec, col, 0,  hh);
      const v16bf b1 = ld_bfrag(sWdec, col, 32, hh);
      v8f a = WMMA_BF16(ha0, b0, zacc);
      a     = WMMA_BF16(ha1, b1, a);
      acc = a;
      return sBdec[col];
    };

    #pragma unroll
    for (int ng = 0; ng < 4; ng++) {
      v8f ai, ag;
      const float bi = dec_gate(0, ng, ai);
      const float bg = dec_gate(2, ng, ag);
      v8f ig;
      #pragma unroll
      for (int e = 0; e < 8; e++)
        ig[e] = fast_sigmoid(ai[e] + bi) * fast_tanh(ag[e] + bg);

      v8f af;
      const float bf_ = dec_gate(1, ng, af);
      #pragma unroll
      for (int e = 0; e < 8; e++)
        cst[ng][e] = fast_sigmoid(af[e] + bf_) * cst[ng][e] + ig[e];

      v8f ao;
      const float bo = dec_gate(3, ng, ao);
      #pragma unroll
      for (int e = 0; e < 8; e++) {
        const float hn = fast_sigmoid(ao[e] + bo) * fast_tanh(cst[ng][e]);
        hbuf[(e + hh * 8) * WS + ng * 16 + rl] = (__bf16)hn;
      }
      __builtin_amdgcn_sched_barrier(0);
    }

    // position head: row rl, dot split across lane halves (j = hh*32 .. +31)
    float p0 = 0.0f, p1 = 0.0f;
    const __bf16* hr = &hbuf[rl * WS + hh * 32];
    const float*  w0 = &sWemb[hh * 32];
    const float*  w1 = &sWemb[HID + hh * 32];
    #pragma unroll
    for (int j = 0; j < 32; j += 2) {
      const unsigned u = *(const unsigned*)(hr + j);
      const float h0 = __uint_as_float(u << 16);
      const float h1 = __uint_as_float(u & 0xffff0000u);
      p0 = fmaf(h0, w0[j], fmaf(h1, w0[j + 1], p0));
      p1 = fmaf(h0, w1[j], fmaf(h1, w1[j + 1], p1));
    }
    p0 += __shfl_xor(p0, 16, 32);
    p1 += __shfl_xor(p1, 16, 32);
    if (hh == 0) {
      float2* op = (float2*)(out + ((size_t)(rowBase + rl) * T_DEC + t) * 2);
      *op = make_float2(p0 + sBemb[0], p1 + sBemb[1]);
    }
  }
}

extern "C" void kernel_launch(void* const* d_in, const int* in_sizes, int n_in,
                              void* d_out, int out_size, void* d_ws, size_t ws_size,
                              hipStream_t stream) {
  const int B = in_sizes[0] / (T_ENC * 2);
  dim3 grid(B / ROWS_PER_BLOCK);
  dim3 block(256);
  lstm_seq2seq_wmma_kernel<<<grid, block, 0, stream>>>(
      (const float*)d_in[0],  (const float*)d_in[1], (const float*)d_in[2],
      (const float*)d_in[3],  (const float*)d_in[4], (const float*)d_in[5],
      (const float*)d_in[6],  (const float*)d_in[7], (const float*)d_in[8],
      (const float*)d_in[9],  (const float*)d_in[10], (float*)d_out);
}
